// GINEncoder_90228672955075
// MI455X (gfx1250) — matrix operand order repaired and checked by reading, not compiled, not run on hardware
//
#include <hip/hip_runtime.h>

// GIN encoder for MI455X (gfx1250, wave32, WMMA).
// Per layer: zero agg -> edge scatter-add (f32 atomics, L2-resident)
// -> pack (x+agg) to f16 -> WMMA GEMM1 (+bias, ReLU, f16 out)
// -> WMMA GEMM2 (+bias, ReLU, f32 out) -> column stats -> batchnorm.
// GEMMs stage the packed weight panel in LDS (double-buffered, async
// global->LDS via ASYNCcnt) so WMMAs are fed from ds_load instead of
// per-instruction global loads.

typedef _Float16 h8   __attribute__((ext_vector_type(8)));
typedef _Float16 v16h __attribute__((ext_vector_type(16)));
typedef float    v8f  __attribute__((ext_vector_type(8)));
typedef int      v4i  __attribute__((ext_vector_type(4)));

typedef __attribute__((address_space(1))) v4i* as1_v4i;
typedef __attribute__((address_space(3))) v4i* as3_v4i;

#define NN 50000
#define FF 128
#define DD 256
#define GG 128
#define EE 800000
#define BN_EPS 1e-5f
#define KC 64            // K-chunk staged per LDS buffer (2 WMMA K-steps)

#if __has_builtin(__builtin_amdgcn_global_load_async_to_lds_b128) && \
    __has_builtin(__builtin_amdgcn_s_wait_asynccnt)
#define USE_ASYNC_LDS 1
#else
#define USE_ASYNC_LDS 0
#endif

__global__ void k_zero(float* __restrict__ p, int n) {
    int i = blockIdx.x * blockDim.x + threadIdx.x;
    if (i < n) p[i] = 0.0f;
}

// One thread handles 8 features of one edge: gather x[src], atomicAdd into agg[dst].
__global__ void k_scatter(const float* __restrict__ x, const int* __restrict__ src,
                          const int* __restrict__ dst, float* __restrict__ agg, int ncols) {
    int t = blockIdx.x * blockDim.x + threadIdx.x;
    int parts = ncols >> 3;
    int e = t / parts;
    if (e >= EE) return;
    int p = t - e * parts;
    int s = src[e], d = dst[e];
    const float4* xs = (const float4*)(x + (size_t)s * ncols + p * 8);
    float4 a = xs[0], b = xs[1];
    float* ad = agg + (size_t)d * ncols + p * 8;
    atomicAdd(ad + 0, a.x); atomicAdd(ad + 1, a.y);
    atomicAdd(ad + 2, a.z); atomicAdd(ad + 3, a.w);
    atomicAdd(ad + 4, b.x); atomicAdd(ad + 5, b.y);
    atomicAdd(ad + 6, b.z); atomicAdd(ad + 7, b.w);
}

// h = x + agg, converted to f16 (row-major [N, ncols]).
__global__ void k_packA(const float* __restrict__ x, const float* __restrict__ agg,
                        _Float16* __restrict__ a16, int n) {
    int i = blockIdx.x * blockDim.x + threadIdx.x;
    if (i < n) a16[i] = (_Float16)(x[i] + agg[i]);
}

// Pack row-major f32 weights [K, Nc] into per-lane WMMA B-fragment order:
// fragment (kt, nt): lane l holds column nt*16+(l&15), Ks kt*32+((l>>4)<<4)+j, j=0..15.
__global__ void k_packB(const float* __restrict__ w, _Float16* __restrict__ bp, int K, int Nc) {
    int i = blockIdx.x * blockDim.x + threadIdx.x;
    if (i >= K * Nc) return;
    int j    = i & 15;
    int lane = (i >> 4) & 31;
    int tile = i >> 9;
    int ntiles = Nc >> 4;
    int nt = tile % ntiles;
    int kt = tile / ntiles;
    int k = kt * 32 + ((lane >> 4) << 4) + j;
    int n = nt * 16 + (lane & 15);
    bp[i] = (_Float16)w[(size_t)k * Nc + n];
}

// Stage one contiguous 32KB B-chunk (KC*DD halves) into LDS with 256 threads.
__device__ __forceinline__ void stage_chunk(const _Float16* __restrict__ gsrc,
                                            _Float16* lbuf, int tid) {
#if USE_ASYNC_LDS
    const char* g = (const char*)gsrc;
    char* l = (char*)lbuf;
#pragma unroll
    for (int i = 0; i < 8; ++i) {
        int off = (tid + i * 256) * 16;
        __builtin_amdgcn_global_load_async_to_lds_b128(
            (as1_v4i)(void*)(g + off),
            (as3_v4i)(void*)(l + off),
            0, 0);
    }
#else
    const uint4* g = (const uint4*)gsrc;
    uint4* l = (uint4*)lbuf;
#pragma unroll
    for (int i = 0; i < 8; ++i) l[tid + i * 256] = g[tid + i * 256];
#endif
}

__device__ __forceinline__ void stage_wait() {
#if USE_ASYNC_LDS
    __builtin_amdgcn_s_wait_asynccnt(0);
#endif
    __syncthreads();
}

// WMMA GEMM: out[N, 256] = relu(A[N, KDIM] @ W[KDIM, 256] + bias).
// 256 threads = 8 waves: 4 row-groups x 2 col-groups; wave tile 16x128 (8 accums).
// B panel double-buffered through LDS in KC-deep chunks.
template <bool F16OUT, int KDIM>
__global__ void k_gemm(const _Float16* __restrict__ A, const _Float16* __restrict__ Bp,
                       const float* __restrict__ bias, void* __restrict__ out) {
    constexpr int NCHUNK = KDIM / KC;
    __shared__ _Float16 sB[2][KC * DD];            // 2 x 32KB

    const int tid  = threadIdx.x;
    const int lane = tid & 31;
    const int wv   = tid >> 5;
    const int rowGroup = wv & 3;
    const int colGroup = wv >> 2;
    const int tileM   = blockIdx.x * 64 + rowGroup * 16;
    const int colBase = colGroup * 128;
    const int col16   = lane & 15;

    int arow = tileM + col16;                      // A row this lane feeds
    if (arow > NN - 1) arow = NN - 1;              // clamp (stores are guarded)
    const int kb = (lane >> 4) * 16;               // 16-bit A fragment K sub-offset

    v8f acc[8];
#pragma unroll
    for (int nt = 0; nt < 8; ++nt) {
        float bv = bias[colBase + nt * 16 + col16];  // C layout: fixed column per lane
        v8f c = {bv, bv, bv, bv, bv, bv, bv, bv};
        acc[nt] = c;
    }

    const _Float16* ap = A + (size_t)arow * KDIM;

    stage_chunk(Bp, &sB[0][0], tid);
    stage_wait();

#pragma unroll
    for (int c = 0; c < NCHUNK; ++c) {
        if (c + 1 < NCHUNK)                          // prefetch next chunk while computing
            stage_chunk(Bp + (size_t)(c + 1) * KC * DD, &sB[(c + 1) & 1][0], tid);

        const _Float16* lb = &sB[c & 1][0];
#pragma unroll
        for (int kk = 0; kk < 2; ++kk) {
            const int k = c * KC + kk * 32;
            union { v16h v; h8 h[2]; } a;
            a.h[0] = *(const h8*)(ap + k + kb);
            a.h[1] = *(const h8*)(ap + k + kb + 8);
#pragma unroll
            for (int nt = 0; nt < 8; ++nt) {
                int ntG = colGroup * 8 + nt;         // global 16-col tile index
                const v16h b = *(const v16h*)(lb + ((size_t)((kk * 16 + ntG) * 32 + lane)) * 16);
                acc[nt] = __builtin_amdgcn_wmma_f32_16x16x32_f16(
                    false, a.v, false, b, (short)0, acc[nt], false, false);
            }
        }
        if (c + 1 < NCHUNK) stage_wait();            // next chunk landed; buffers safe to swap
    }

#pragma unroll
    for (int nt = 0; nt < 8; ++nt) {
        int col = colBase + nt * 16 + col16;
#pragma unroll
        for (int r = 0; r < 8; ++r) {
            int row = tileM + r + ((lane >> 4) << 3);
            if (row < NN) {
                float v = acc[nt][r];
                v = v > 0.0f ? v : 0.0f;             // fused ReLU
                if (F16OUT) ((_Float16*)out)[(size_t)row * DD + col] = (_Float16)v;
                else        ((float*)out)[(size_t)row * DD + col] = v;
            }
        }
    }
}

// Per-column sum / sum-of-squares over a 256-row chunk; one thread per column.
__global__ void k_colstats(const float* __restrict__ h, float* __restrict__ sum,
                           float* __restrict__ sumsq) {
    int col = threadIdx.x;            // blockDim.x == 256 == DD
    int r0 = blockIdx.x * 256;
    int rend = r0 + 256; if (rend > NN) rend = NN;
    float s = 0.0f, q = 0.0f;
    for (int r = r0; r < rend; ++r) {
        float v = h[(size_t)r * DD + col];
        s += v; q += v * v;
    }
    atomicAdd(&sum[col], s);
    atomicAdd(&sumsq[col], q);
}

__global__ void k_bn(const float* __restrict__ h, const float* __restrict__ sum,
                     const float* __restrict__ sumsq, const float* __restrict__ gamma,
                     const float* __restrict__ beta, float* __restrict__ xo, int n) {
    int i = blockIdx.x * blockDim.x + threadIdx.x;
    if (i >= n) return;
    int col = i & (DD - 1);
    const float invN = 1.0f / (float)NN;
    float mean = sum[col] * invN;
    float var  = sumsq[col] * invN - mean * mean;
    float inv  = rsqrtf(var + BN_EPS);
    xo[i] = gamma[col] * (h[i] - mean) * inv + beta[col];
}

// global_add_pool: one thread per (node, 4-feature chunk).
__global__ void k_pool(const float* __restrict__ x, const int* __restrict__ batch,
                       float* __restrict__ out) {
    int t = blockIdx.x * blockDim.x + threadIdx.x;
    if (t >= NN * 64) return;
    int node = t >> 6;
    int c4 = (t & 63) << 2;
    int g = batch[node];
    float4 v = *(const float4*)(x + (size_t)node * DD + c4);
    float* o = out + (size_t)g * DD + c4;
    atomicAdd(o + 0, v.x); atomicAdd(o + 1, v.y);
    atomicAdd(o + 2, v.z); atomicAdd(o + 3, v.w);
}

extern "C" void kernel_launch(void* const* d_in, const int* in_sizes, int n_in,
                              void* d_out, int out_size, void* d_ws, size_t ws_size,
                              hipStream_t stream) {
    (void)in_sizes; (void)n_in; (void)out_size; (void)ws_size;

    const float* x0    = (const float*)d_in[0];
    const int*   ei    = (const int*)d_in[1];
    const int*   src   = ei;
    const int*   dst   = ei + EE;
    const int*   batch = (const int*)d_in[2];

    // Workspace carve-out (~205 MB total)
    char* ws = (char*)d_ws;
    size_t off = 0;
    auto take = [&](size_t bytes) -> char* {
        char* p = ws + off;
        off = (off + bytes + 255) & ~(size_t)255;
        return p;
    };
    float*    x_cur = (float*)take((size_t)NN * DD * 4);
    float*    agg   = (float*)take((size_t)NN * DD * 4);
    float*    h2    = (float*)take((size_t)NN * DD * 4);
    _Float16* a16   = (_Float16*)take((size_t)NN * DD * 2);
    _Float16* hm16  = (_Float16*)take((size_t)NN * DD * 2);
    _Float16* w1p   = (_Float16*)take((size_t)DD * DD * 2);
    _Float16* w2p   = (_Float16*)take((size_t)DD * DD * 2);
    float*    stats = (float*)take((size_t)2 * DD * 4);

    for (int l = 0; l < 3; ++l) {
        int fin = (l == 0) ? FF : DD;
        const float* xin   = (l == 0) ? x0 : x_cur;
        const float* w1    = (const float*)d_in[3 + 6 * l + 0];
        const float* b1    = (const float*)d_in[3 + 6 * l + 1];
        const float* w2    = (const float*)d_in[3 + 6 * l + 2];
        const float* b2    = (const float*)d_in[3 + 6 * l + 3];
        const float* gamma = (const float*)d_in[3 + 6 * l + 4];
        const float* beta  = (const float*)d_in[3 + 6 * l + 5];

        int nagg = NN * fin;
        k_zero<<<(nagg + 255) / 256, 256, 0, stream>>>(agg, nagg);

        int nsc = EE * (fin / 8);
        k_scatter<<<(nsc + 255) / 256, 256, 0, stream>>>(xin, src, dst, agg, fin);

        k_packA<<<(nagg + 255) / 256, 256, 0, stream>>>(xin, agg, a16, nagg);
        k_packB<<<(fin * DD + 255) / 256, 256, 0, stream>>>(w1, w1p, fin, DD);
        k_packB<<<(DD * DD + 255) / 256, 256, 0, stream>>>(w2, w2p, DD, DD);

        if (l == 0)
            k_gemm<true, FF><<<(NN + 63) / 64, 256, 0, stream>>>(a16, w1p, b1, hm16);
        else
            k_gemm<true, DD><<<(NN + 63) / 64, 256, 0, stream>>>(a16, w1p, b1, hm16);

        k_zero<<<1, 2 * DD, 0, stream>>>(stats, 2 * DD);
        k_gemm<false, DD><<<(NN + 63) / 64, 256, 0, stream>>>(hm16, w2p, b2, h2);

        k_colstats<<<(NN + 255) / 256, 256, 0, stream>>>(h2, stats, stats + DD);
        k_bn<<<(NN * DD + 255) / 256, 256, 0, stream>>>(h2, stats, stats + DD,
                                                        gamma, beta, x_cur, NN * DD);
    }

    k_zero<<<(GG * DD + 255) / 256, 256, 0, stream>>>((float*)d_out, GG * DD);
    k_pool<<<(NN * 64 + 255) / 256, 256, 0, stream>>>(x_cur, batch, (float*)d_out);
}
